// BsplineKAN_54082228191464
// MI455X (gfx1250) — compile-verified, matrix operand order checked by reading, not run on hardware
//
#include <hip/hip_runtime.h>

// ---------------------------------------------------------------------------
// B-spline KAN layer for MI455X (gfx1250, wave32, WMMA)
//   out = LN( einsum(basis(x), cp) + x@W.T + b )
// v4: B-tile copied with GLOBAL_LOAD_ASYNC_TO_LDS_B128 (ASYNCcnt-tracked,
//     bypasses VGPRs), correctly typed per clang's builtin signature.
//     Sparse 4-wide basis scatter, 128x256 bf16 WMMA GEMM, bias, LayerNorm.
// ---------------------------------------------------------------------------

#define BATCH   16384
#define IN_F    1024
#define OUT_F   1024
#define NB      11          // spline basis columns (K in reference)
#define KE      12          // extended: 11 basis + raw x (folds in x@W.T)
#define RED     (IN_F * KE) // 12288 reduction length
#define FB      8           // input features per GEMM iteration
#define MT      128         // M tile
#define NT      256         // N tile
#define LDA     104         // LDS row stride (ushorts): 96 + 8 pad (208B = 52 dwords, conflict-free)
#define EPS     1e-5f

#if __has_builtin(__builtin_amdgcn_global_load_async_to_lds_b128)
#define USE_ASYNC_LDS 1
#else
#define USE_ASYNC_LDS 0
#endif

typedef __attribute__((ext_vector_type(16))) __bf16 v16bf;
typedef __attribute__((ext_vector_type(8)))  float  v8f;
typedef int v4i __attribute__((vector_size(16)));            // matches builtin param
typedef __attribute__((address_space(1))) v4i* gv4i_p;       // global src
typedef __attribute__((address_space(3))) v4i* lv4i_p;       // LDS dst

union Frag32 {
    uint4 u[2];
    v16bf v;
};

static __device__ __forceinline__ unsigned short f2bf(float f) {
    return __builtin_bit_cast(unsigned short, static_cast<__bf16>(f));
}

static __device__ __forceinline__ void wait_async_zero() {
#if __has_builtin(__builtin_amdgcn_s_wait_asynccnt)
    __builtin_amdgcn_s_wait_asynccnt(0);
#else
    asm volatile("s_wait_asynccnt 0x0" ::: "memory");
#endif
}

// ---------------------------------------------------------------------------
// Kernel 1: pack [control_points | W] as bf16 into Bfull[OUT_F][IN_F*12]
//   Bfull[o][i*12 + k] = cp[o][i][k] (k<11),  Bfull[o][i*12+11] = W[o][i]
// ---------------------------------------------------------------------------
__global__ __launch_bounds__(256) void kan_pack_b(
        const float* __restrict__ cp, const float* __restrict__ W,
        unsigned short* __restrict__ Bfull) {
    const int idx = blockIdx.x * 256 + threadIdx.x;   // (o, i) flat
    const float* src = cp + (size_t)idx * NB;
    unsigned short h[KE];
#pragma unroll
    for (int k = 0; k < NB; ++k) h[k] = f2bf(src[k]);
    h[NB] = f2bf(W[idx]);

    unsigned short* dst = Bfull + (size_t)idx * KE;   // 24B per pair, 8B aligned
#pragma unroll
    for (int k = 0; k < KE; k += 4) {
        uint2 w;
        w.x = (unsigned int)h[k]     | ((unsigned int)h[k + 1] << 16);
        w.y = (unsigned int)h[k + 2] | ((unsigned int)h[k + 3] << 16);
        *(uint2*)(dst + k) = w;
    }
}

// ---------------------------------------------------------------------------
// Sparse Cox-de Boor: only columns {jh-3..jh} (jh = floor(11x)) can be
// nonzero, including the reference's truncated-update rule (col j updated at
// degree d iff j < 11-d).  v[k] = value of column (jh-k).
//   alpha_j = (11x - j)/d = (f+k)/d,   beta_j = 1 - alpha_{j+1}
// ---------------------------------------------------------------------------
static __device__ __forceinline__ void basis_sparse(float x, float v[4], int& jh) {
    const float u  = x * 11.0f;
    const float jf = floorf(u);
    jh = (int)jf;
    const float f = u - jf;                        // in [0,1)
    v[0] = (x >= 0.0f && x < 1.0f) ? 1.0f : 0.0f;  // degree-0 indicator at jh
    v[1] = 0.0f; v[2] = 0.0f; v[3] = 0.0f;
#pragma unroll
    for (int d = 1; d <= 3; ++d) {
        const float invd  = 1.0f / (float)d;
        const float abase = f * invd;
#pragma unroll
        for (int k = d; k >= 0; --k) {            // j ascending => old v[k-1]
            float a  = abase + (float)k * invd;               // alpha_{jh-k}
            float b  = (float)(d + 1 - k) * invd - abase;     // 1 - alpha_{jh-k+1}
            float pv = (k > 0) ? v[k - 1] : 0.0f;             // old bas[j+1]
            float nv = a * v[k] + b * pv;
            v[k] = (jh < 11 - d + k) ? nv : v[k];             // truncation rule
        }
    }
}

// ---------------------------------------------------------------------------
// Kernel 2: fused GEMM. Grid (128 mblocks, 4 nblocks), 256 threads = 8 waves.
// Workgroup tile 128x256; wave (wm,wn) owns 64x64 = 4x4 wmma 16x16 tiles.
// ---------------------------------------------------------------------------
__global__ __launch_bounds__(256) void kan_gemm(
        const float* __restrict__ x, const unsigned short* __restrict__ Bfull,
        const float* __restrict__ bias, float* __restrict__ out) {
    __shared__ __attribute__((aligned(16))) unsigned short lA[MT * LDA];
    __shared__ __attribute__((aligned(16))) unsigned short lB[NT * LDA];

    const int tid  = threadIdx.x;
    const int wave = tid >> 5;
    const int lane = tid & 31;
    const int hf   = lane >> 4;   // half-wave: selects K sub-groups
    const int r    = lane & 15;   // row/col within 16

    const int row0 = blockIdx.x * MT;
    const int col0 = blockIdx.y * NT;
    const int wm = wave & 1;      // 0..1  -> 64-row half
    const int wn = wave >> 1;     // 0..3  -> 64-col quarter

    // loader roles
    const int lm = tid >> 1;            // x row 0..127 (2 threads/row)
    const int lf = (tid & 1) * 4;       // 4 features each
    // B: one output-channel row per thread (256 rows, 96 bf16 each)

    v8f acc[4][4];
#pragma unroll
    for (int a = 0; a < 4; ++a)
#pragma unroll
        for (int b = 0; b < 4; ++b)
            acc[a][b] = (v8f){0.f,0.f,0.f,0.f,0.f,0.f,0.f,0.f};

    const float* xrow = x + (size_t)(row0 + lm) * IN_F + lf;
    const unsigned short* brow = Bfull + (size_t)(col0 + tid) * RED;

    for (int ib = 0; ib < IN_F; ib += FB) {
        // ---- stage x into registers; pre-touch next B tile ----
        float4 xv = *(const float4*)(xrow + ib);
        if (ib + FB < IN_F)  // next B tile into caches (global_prefetch_b8)
            __builtin_prefetch((const void*)(brow + (size_t)(ib + FB) * KE), 0, 1);

#if !USE_ASYNC_LDS
        const uint4* gbs = (const uint4*)(brow + (size_t)ib * KE);
        uint4 breg[12];
#pragma unroll
        for (int e = 0; e < 12; ++e) breg[e] = gbs[e];
#endif

        __syncthreads();   // all waves done reading LDS of previous iter

        // ---- B tile: 1 row/thread, 12 x b128, global -> LDS ----
#if USE_ASYNC_LDS
        {
            const char* gsrc = (const char*)(brow + (size_t)ib * KE);
            unsigned short* ldst = &lB[tid * LDA];
#pragma unroll
            for (int e = 0; e < 12; ++e)
                __builtin_amdgcn_global_load_async_to_lds_b128(
                    (gv4i_p)(gsrc + e * 16),
                    (lv4i_p)(ldst + e * 8),   // 8 ushorts = 16 bytes
                    0, 0);
        }
#else
        {
            uint4* lb = (uint4*)&lB[tid * LDA];  // 16B aligned (208B stride)
#pragma unroll
            for (int e = 0; e < 12; ++e) lb[e] = breg[e];
        }
#endif

        // ---- A tile: sparse basis scatter (overlaps async B copy) ----
        {
            float xs[4] = {xv.x, xv.y, xv.z, xv.w};
#pragma unroll
            for (int q = 0; q < 4; ++q) {
                float v[4]; int jh;
                basis_sparse(xs[q], v, jh);
                unsigned short* pa = &lA[lm * LDA + (lf + q) * KE];  // 8B aligned
                // zero 12 cols, with bf16(x) pre-packed into col 11
                uint2 z; z.x = 0u; z.y = 0u;
                *(uint2*)(pa + 0) = z;                               // cols 0..3
                *(uint2*)(pa + 4) = z;                               // cols 4..7
                z.y = ((unsigned int)f2bf(xs[q])) << 16;             // col 11 = x
                *(uint2*)(pa + 8) = z;                               // cols 8..11
                // scatter the (at most) 4 nonzero basis values
#pragma unroll
                for (int k = 0; k < 4; ++k) {
                    int col = jh - k;
                    if (col >= 0 && col <= 10) pa[col] = f2bf(v[k]);
                }
            }
        }

#if USE_ASYNC_LDS
        wait_async_zero();   // this wave's async LDS writes done
#endif
        __syncthreads();

        // ---- 3 k-steps of 32, 16 wmma each ----
#pragma unroll
        for (int s = 0; s < 3; ++s) {
            Frag32 fa[4];
#pragma unroll
            for (int a = 0; a < 4; ++a) {
                const unsigned short* pa =
                    &lA[(wm * 64 + a * 16 + r) * LDA + s * 32 + hf * 8];
                fa[a].u[0] = *(const uint4*)pa;        // K = s*32 + hf*8 + 0..7
                fa[a].u[1] = *(const uint4*)(pa + 16); // K = s*32 + 16 + hf*8 + 0..7
            }
#pragma unroll
            for (int b = 0; b < 4; ++b) {
                Frag32 fb;
                const unsigned short* pb =
                    &lB[(wn * 64 + b * 16 + r) * LDA + s * 32 + hf * 8];
                fb.u[0] = *(const uint4*)pb;
                fb.u[1] = *(const uint4*)(pb + 16);
#pragma unroll
                for (int a = 0; a < 4; ++a)
                    acc[a][b] = __builtin_amdgcn_wmma_f32_16x16x32_bf16(
                        false, fa[a].v, false, fb.v,
                        (short)0, acc[a][b], false, false);
            }
        }
    }

    // ---- epilogue: + bias, store fp32 per C/D layout (M = v + 8*hf, N = r) ----
#pragma unroll
    for (int b = 0; b < 4; ++b) {
        const int col = col0 + wn * 64 + b * 16 + r;
        const float bv = bias[col];
#pragma unroll
        for (int a = 0; a < 4; ++a) {
            const int rbase = row0 + wm * 64 + a * 16 + hf * 8;
#pragma unroll
            for (int v = 0; v < 8; ++v)
                out[(size_t)(rbase + v) * OUT_F + col] = acc[a][b][v] + bv;
        }
    }
}

// ---------------------------------------------------------------------------
// Kernel 3: LayerNorm over last dim, in place on d_out. 1 block = 1 row.
// ---------------------------------------------------------------------------
__global__ __launch_bounds__(256) void kan_ln(
        float* __restrict__ out, const float* __restrict__ gamma,
        const float* __restrict__ beta) {
    __shared__ float rsum[256];
    __shared__ float rsq[256];
    const int tid = threadIdx.x;
    float* p = out + (size_t)blockIdx.x * OUT_F;

    float4 v = ((const float4*)p)[tid];   // 256 * 4 = 1024
    float s = v.x + v.y + v.z + v.w;
    float q = v.x * v.x + v.y * v.y + v.z * v.z + v.w * v.w;
    rsum[tid] = s;
    rsq[tid]  = q;
    __syncthreads();
#pragma unroll
    for (int off = 128; off > 0; off >>= 1) {
        if (tid < off) {
            rsum[tid] += rsum[tid + off];
            rsq[tid]  += rsq[tid + off];
        }
        __syncthreads();
    }
    const float mean = rsum[0] * (1.0f / OUT_F);
    const float var  = rsq[0] * (1.0f / OUT_F) - mean * mean;
    const float inv  = rsqrtf(var + EPS);

    float4 g  = ((const float4*)gamma)[tid];
    float4 be = ((const float4*)beta)[tid];
    v.x = g.x * (v.x - mean) * inv + be.x;
    v.y = g.y * (v.y - mean) * inv + be.y;
    v.z = g.z * (v.z - mean) * inv + be.z;
    v.w = g.w * (v.w - mean) * inv + be.w;
    ((float4*)p)[tid] = v;
}

// ---------------------------------------------------------------------------
extern "C" void kernel_launch(void* const* d_in, const int* in_sizes, int n_in,
                              void* d_out, int out_size, void* d_ws, size_t ws_size,
                              hipStream_t stream) {
    const float* x     = (const float*)d_in[0];   // [16384,1024]
    const float* cp    = (const float*)d_in[1];   // [1024,1024,11]
    const float* W     = (const float*)d_in[2];   // [1024,1024]
    const float* bias  = (const float*)d_in[3];   // [1024]
    const float* gamma = (const float*)d_in[4];   // [1024]
    const float* beta  = (const float*)d_in[5];   // [1024]
    float* out = (float*)d_out;                   // [16384,1024] fp32

    unsigned short* Bfull = (unsigned short*)d_ws; // OUT_F*RED*2 = 25,165,824 B

    kan_pack_b<<<(OUT_F * IN_F) / 256, 256, 0, stream>>>(cp, W, Bfull);
    kan_gemm<<<dim3(BATCH / MT, OUT_F / NT), 256, 0, stream>>>(x, Bfull, bias, out);
    kan_ln<<<BATCH, 256, 0, stream>>>(out, gamma, beta);
}
